// NodeEdgeCrossAttention_70789650972749
// MI455X (gfx1250) — matrix operand order repaired and checked by reading, not compiled, not run on hardware
//
#include <hip/hip_runtime.h>

#define N_NODES 50000
#define N_EDGES 640000
#define DIM     128
#define HEADS   4
#define D_HEAD  32
#define SCALE   0.17677669529663687f  // 32^-0.5

typedef __attribute__((ext_vector_type(16))) _Float16 v16h;
typedef __attribute__((ext_vector_type(8)))  float    v8f;

// ---------- order-preserving float <-> u32 key (for atomic segment-max) ----------
__device__ __forceinline__ unsigned fkey(float f) {
    unsigned b = __float_as_uint(f);
    return (b & 0x80000000u) ? ~b : (b | 0x80000000u);
}
__device__ __forceinline__ float funkey(unsigned k) {
    unsigned b = (k & 0x80000000u) ? (k & 0x7fffffffu) : ~k;
    return __uint_as_float(b);
}

// ---------- zero init ----------
__global__ void zero_u32_kernel(unsigned* __restrict__ p, size_t n) {
    size_t i = (size_t)blockIdx.x * blockDim.x + threadIdx.x;
    if (i < n) p[i] = 0u;
}

// ---------- pack W[128,128] f32 into per-lane WMMA B fragments (f16) ----------
// P[((t*4 + c)*32 + L)*16 + e] = W[(32c + 16*(L>>4) + e)*128 + 16t + (L&15)]
// t = column tile (0..7), c = K chunk (0..3), L = lane, e = fragment element.
__global__ __launch_bounds__(256) void pack_w_kernel(
    const float* __restrict__ W, _Float16* __restrict__ P) {
    const int tid = blockIdx.x * blockDim.x + threadIdx.x;   // 0..1023
    if (tid >= 1024) return;
    const int t    = tid >> 7;
    const int c    = (tid >> 5) & 3;
    const int L    = tid & 31;
    const int half = L >> 4;
    const int lm   = L & 15;
    _Float16* p = P + (size_t)tid * 16;
#pragma unroll
    for (int e = 0; e < 16; ++e)
        p[e] = (_Float16)W[(size_t)(32 * c + 16 * half + e) * DIM + 16 * t + lm];
}

// ---------- Y[M,128] = X[M,128] @ W + b  via v_wmma_f32_16x16x32_f16 ----------
// Block: 256 threads = 8 waves; block covers 16 rows, wave w covers cols [16w,16w+16).
// Wp is the pre-packed f16 B-fragment buffer. gridDim.x = M/16.
template <typename OutT>
__global__ __launch_bounds__(256) void wmma_gemm128_kernel(
    const float* __restrict__ X, const _Float16* __restrict__ Wp,
    const float* __restrict__ bias, OutT* __restrict__ Y) {
    const int lane = threadIdx.x & 31;
    const int wave = threadIdx.x >> 5;   // 0..7 -> column tile
    const int half = lane >> 4;          // 0/1
    const int lm   = lane & 15;
    const size_t rowBase = (size_t)blockIdx.x * 16;
    const int colBase    = wave * 16;

    const float*    xrow = X + (rowBase + lm) * DIM;                       // A row
    const _Float16* wp   = Wp + (((size_t)wave * 4) * 32 + lane) * 16;     // B frags

    v8f acc = {};
#pragma unroll
    for (int c = 0; c < 4; ++c) {
        const int kb = c * 32;
        // A (16x32 f16): lane (half,lm): K = half*16+0..15 + kb
        const float4 p0 = *(const float4*)(xrow + kb + half * 16);
        const float4 p1 = *(const float4*)(xrow + kb + half * 16 + 4);
        const float4 p2 = *(const float4*)(xrow + kb + half * 16 + 8);
        const float4 p3 = *(const float4*)(xrow + kb + half * 16 + 12);
        v16h a;
        a[0]  = (_Float16)p0.x; a[1]  = (_Float16)p0.y; a[2]  = (_Float16)p0.z; a[3]  = (_Float16)p0.w;
        a[4]  = (_Float16)p1.x; a[5]  = (_Float16)p1.y; a[6]  = (_Float16)p1.z; a[7]  = (_Float16)p1.w;
        a[8]  = (_Float16)p2.x; a[9]  = (_Float16)p2.y; a[10] = (_Float16)p2.z; a[11] = (_Float16)p2.w;
        a[12] = (_Float16)p3.x; a[13] = (_Float16)p3.y; a[14] = (_Float16)p3.z; a[15] = (_Float16)p3.w;
        // B fragment: one aligned 32-byte vector load, already f16
        const v16h bb = *(const v16h*)(wp + (size_t)c * 32 * 16);
        acc = __builtin_amdgcn_wmma_f32_16x16x32_f16(
            /*neg_a=*/false, a, /*neg_b=*/false, bb,
            /*c_mod=*/(short)0, acc, /*reuse_a=*/false, /*reuse_b=*/false);
    }
    // D layout: acc[j] -> row rowBase + j + 8*half, col colBase + lm
    const float bv = bias[colBase + lm];
#pragma unroll
    for (int j = 0; j < 8; ++j) {
        const size_t r = rowBase + (size_t)j + 8 * half;
        Y[r * DIM + colBase + lm] = (OutT)(acc[j] + bv);
    }
}

// ---------- per-edge attention score + keyed segment max (wave per edge) ----------
__global__ __launch_bounds__(256) void score_kernel(
    const float* __restrict__ qp, const _Float16* __restrict__ kp,
    const int* __restrict__ dst, float* __restrict__ score,
    unsigned* __restrict__ segkey) {
    const int wid  = blockIdx.x * 8 + (threadIdx.x >> 5);
    const int lane = threadIdx.x & 31;
    if (wid >= N_EDGES) return;
    const int d = dst[wid];
    const float*    q = qp + (size_t)d * DIM;
    const _Float16* k = kp + (size_t)wid * DIM;
    float s0 = q[lane]            * (float)k[lane];
    float s1 = q[D_HEAD   + lane] * (float)k[D_HEAD   + lane];
    float s2 = q[2*D_HEAD + lane] * (float)k[2*D_HEAD + lane];
    float s3 = q[3*D_HEAD + lane] * (float)k[3*D_HEAD + lane];
#pragma unroll
    for (int off = 16; off >= 1; off >>= 1) {
        s0 += __shfl_xor(s0, off, 32);
        s1 += __shfl_xor(s1, off, 32);
        s2 += __shfl_xor(s2, off, 32);
        s3 += __shfl_xor(s3, off, 32);
    }
    if (lane == 0) {
        s0 *= SCALE; s1 *= SCALE; s2 *= SCALE; s3 *= SCALE;
        score[(size_t)wid * 4 + 0] = s0; atomicMax(&segkey[(size_t)d * 4 + 0], fkey(s0));
        score[(size_t)wid * 4 + 1] = s1; atomicMax(&segkey[(size_t)d * 4 + 1], fkey(s1));
        score[(size_t)wid * 4 + 2] = s2; atomicMax(&segkey[(size_t)d * 4 + 2], fkey(s2));
        score[(size_t)wid * 4 + 3] = s3; atomicMax(&segkey[(size_t)d * 4 + 3], fkey(s3));
    }
}

// ---------- ex = exp(score - segmax[dst]); denom += ex (in-place over score) ----------
__global__ __launch_bounds__(256) void softmax_num_kernel(
    float* __restrict__ score, const int* __restrict__ dst,
    const unsigned* __restrict__ segkey, float* __restrict__ denom) {
    const size_t i = (size_t)blockIdx.x * 256 + threadIdx.x;
    if (i >= (size_t)N_EDGES * HEADS) return;
    const int e = (int)(i >> 2);
    const int h = (int)(i & 3);
    const int d = dst[e];
    const float m  = funkey(segkey[(size_t)d * 4 + h]);
    const float ex = __expf(score[i] - m);
    score[i] = ex;
    atomicAdd(&denom[(size_t)d * 4 + h], ex);
}

// ---------- weighted-V scatter to nodes (wave per edge) ----------
__global__ __launch_bounds__(256) void scatter_kernel(
    const _Float16* __restrict__ vp, const float* __restrict__ ex,
    const float* __restrict__ denom, const int* __restrict__ dst,
    float* __restrict__ agg) {
    const int wid  = blockIdx.x * 8 + (threadIdx.x >> 5);
    const int lane = threadIdx.x & 31;
    if (wid >= N_EDGES) return;
    const int d = dst[wid];
#pragma unroll
    for (int h = 0; h < HEADS; ++h) {
        const float a = ex[(size_t)wid * 4 + h] / denom[(size_t)d * 4 + h];
        const float w = (float)vp[(size_t)wid * DIM + h * D_HEAD + lane] * a;
        atomicAdd(&agg[(size_t)d * DIM + h * D_HEAD + lane], w);
    }
}

extern "C" void kernel_launch(void* const* d_in, const int* in_sizes, int n_in,
                              void* d_out, int out_size, void* d_ws, size_t ws_size,
                              hipStream_t stream) {
    (void)in_sizes; (void)n_in; (void)out_size; (void)ws_size;
    const float* q_nodes = (const float*)d_in[0];
    const float* k_edges = (const float*)d_in[1];
    const float* v_edges = (const float*)d_in[2];
    const int*   dst     = (const int*)d_in[3];     // edge_index[0] = first E entries
    const float* Wq = (const float*)d_in[4];  const float* bq = (const float*)d_in[5];
    const float* Wk = (const float*)d_in[6];  const float* bk = (const float*)d_in[7];
    const float* Wv = (const float*)d_in[8];  const float* bv = (const float*)d_in[9];
    const float* Wo = (const float*)d_in[10]; const float* bo = (const float*)d_in[11];
    float* out = (float*)d_out;

    // ---- workspace carve-out (section sizes are 256B multiples) ----
    char* ws = (char*)d_ws;
    size_t off = 0;
    auto carve = [&](size_t bytes) -> void* {
        void* p = ws + off;
        off += (bytes + 255) & ~(size_t)255;
        return p;
    };
    // zeroed accumulators placed contiguously so one kernel clears all three
    unsigned* segkey = (unsigned*)carve((size_t)N_NODES * 4 * sizeof(unsigned));
    float*    denom  = (float*)   carve((size_t)N_NODES * 4 * sizeof(float));
    float*    agg    = (float*)   carve((size_t)N_NODES * DIM * sizeof(float));
    float*    qp     = (float*)   carve((size_t)N_NODES * DIM * sizeof(float));
    float*    score  = (float*)   carve((size_t)N_EDGES * HEADS * sizeof(float)); // reused as ex
    _Float16* kp     = (_Float16*)carve((size_t)N_EDGES * DIM * sizeof(_Float16));
    _Float16* vp     = (_Float16*)carve((size_t)N_EDGES * DIM * sizeof(_Float16));
    _Float16* Wqp    = (_Float16*)carve((size_t)DIM * DIM * sizeof(_Float16));
    _Float16* Wkp    = (_Float16*)carve((size_t)DIM * DIM * sizeof(_Float16));
    _Float16* Wvp    = (_Float16*)carve((size_t)DIM * DIM * sizeof(_Float16));
    _Float16* Wop    = (_Float16*)carve((size_t)DIM * DIM * sizeof(_Float16));

    // 1) zero segkey/denom/agg (contiguous 136*N u32 words)
    const size_t nzero = (size_t)N_NODES * (4 + 4 + DIM);
    zero_u32_kernel<<<(unsigned)((nzero + 255) / 256), 256, 0, stream>>>(segkey, nzero);

    // 2) pack the four weight matrices into f16 WMMA B-fragment layout (once)
    pack_w_kernel<<<4, 256, 0, stream>>>(Wq, Wqp);
    pack_w_kernel<<<4, 256, 0, stream>>>(Wk, Wkp);
    pack_w_kernel<<<4, 256, 0, stream>>>(Wv, Wvp);
    pack_w_kernel<<<4, 256, 0, stream>>>(Wo, Wop);

    // 3) projections (WMMA f16 -> f32 accumulate)
    wmma_gemm128_kernel<float>   <<<N_NODES / 16, 256, 0, stream>>>(q_nodes, Wqp, bq, qp);
    wmma_gemm128_kernel<_Float16><<<N_EDGES / 16, 256, 0, stream>>>(k_edges, Wkp, bk, kp);
    wmma_gemm128_kernel<_Float16><<<N_EDGES / 16, 256, 0, stream>>>(v_edges, Wvp, bv, vp);

    // 4) per-edge scores + segment max
    score_kernel<<<N_EDGES / 8, 256, 0, stream>>>(qp, kp, dst, score, segkey);

    // 5) softmax numerator + segment sum
    softmax_num_kernel<<<(N_EDGES * HEADS) / 256, 256, 0, stream>>>(score, dst, segkey, denom);

    // 6) weighted-V scatter-sum to nodes
    scatter_kernel<<<N_EDGES / 8, 256, 0, stream>>>(vp, score, denom, dst, agg);

    // 7) output projection straight into d_out
    wmma_gemm128_kernel<float><<<N_NODES / 16, 256, 0, stream>>>(agg, Wop, bo, out);
}
